// GCNLinkPredictor_22703197127227
// MI455X (gfx1250) — matrix operand order repaired and checked by reading, not compiled, not run on hardware
//
#include <hip/hip_runtime.h>
#include <hip/hip_bf16.h>

#define NN 100000
#define EE 3200000
#define FF 128
#define UU 1000

typedef __attribute__((ext_vector_type(16))) __bf16 v16bf;
typedef __attribute__((ext_vector_type(8)))  __bf16 v8bf;
typedef __attribute__((ext_vector_type(8)))  float  v8f;

// native f32 -> bf16 conversion (RNE); lets clang emit v_cvt_*bf16* ops
static __device__ __forceinline__ __bf16 f2bf(float f) {
    return (__bf16)f;
}

// ---------------- degree / normalization ----------------
__global__ void fill_ones(float* __restrict__ p, int n) {
    int i = blockIdx.x * blockDim.x + threadIdx.x;
    if (i < n) p[i] = 1.0f;
}

__global__ void deg_count(const long long* __restrict__ dst, float* __restrict__ deg, int e) {
    int i = blockIdx.x * blockDim.x + threadIdx.x;
    if (i < e) atomicAdd(&deg[(int)dst[i]], 1.0f);
}

__global__ void rsqrt_ip(float* __restrict__ p, int n) {
    int i = blockIdx.x * blockDim.x + threadIdx.x;
    if (i < n) p[i] = rsqrtf(p[i]);
}

// ---------------- W -> bf16, transposed (col-major) ----------------
// Wt[n*FF + k] = bf16(W[k*FF + n]) so a B-fragment is contiguous in k.
__global__ void conv_wt(const float* __restrict__ W, __bf16* __restrict__ Wt) {
    int i = blockIdx.x * blockDim.x + threadIdx.x;
    if (i < FF * FF) {
        int n = i >> 7, k = i & 127;
        Wt[i] = f2bf(W[k * FF + n]);
    }
}

// ---------------- WMMA GEMM: Y[M,F] = X[M,F] @ W (Wt = W^T bf16) ----------------
// One wave per 16-row tile. 8 N-tiles x 4 K-chunks = 32 v_wmma_f32_16x16x32_bf16.
__global__ void gemm_wmma(const float* __restrict__ X, const __bf16* __restrict__ Wt,
                          float* __restrict__ Y, int nrows) {
    int wid  = (blockIdx.x * blockDim.x + threadIdx.x) >> 5;
    int lane = threadIdx.x & 31;
    int m0 = wid << 4;
    if (m0 >= nrows) return;              // uniform per wave -> EXEC all ones at WMMA
    int m    = lane & 15;                 // A: row index / B: col index / C: col index
    int half = lane >> 4;

    // A fragments: a[kc][i] = A[m][kc*32 + 16*(i/8) + 8*half + (i%8)]  (ISA 16-bit A layout)
    const float* xrow = X + (size_t)(m0 + m) * FF;
    v16bf a[4];
#pragma unroll
    for (int kc = 0; kc < 4; ++kc) {
        const float* p = xrow + kc * 32 + half * 8;
        float4 f0 = *(const float4*)(p);
        float4 f1 = *(const float4*)(p + 4);
        float4 f2 = *(const float4*)(p + 16);
        float4 f3 = *(const float4*)(p + 20);
        v16bf t;
        t[0]  = f2bf(f0.x); t[1]  = f2bf(f0.y); t[2]  = f2bf(f0.z); t[3]  = f2bf(f0.w);
        t[4]  = f2bf(f1.x); t[5]  = f2bf(f1.y); t[6]  = f2bf(f1.z); t[7]  = f2bf(f1.w);
        t[8]  = f2bf(f2.x); t[9]  = f2bf(f2.y); t[10] = f2bf(f2.z); t[11] = f2bf(f2.w);
        t[12] = f2bf(f3.x); t[13] = f2bf(f3.y); t[14] = f2bf(f3.z); t[15] = f2bf(f3.w);
        a[kc] = t;
    }

#pragma unroll
    for (int nt = 0; nt < 8; ++nt) {
        int n0 = nt << 4;
        v8f acc = {};
#pragma unroll
        for (int kc = 0; kc < 4; ++kc) {
            // B fragment: b[i] = W[kc*32 + 16*(i/8) + 8*half + (i%8)][n0+m] = Wt[(n0+m)*FF + kc*32 + 16*(i/8) + 8*half + (i%8)]
            const __bf16* wp = Wt + (size_t)(n0 + m) * FF + kc * 32 + half * 8;
            v8bf b0 = *(const v8bf*)(wp);
            v8bf b1 = *(const v8bf*)(wp + 16);
            v16bf b;
#pragma unroll
            for (int i = 0; i < 8; ++i) { b[i] = b0[i]; b[i + 8] = b1[i]; }
            acc = __builtin_amdgcn_wmma_f32_16x16x32_bf16(
                false, a[kc], false, b, (short)0, acc, false, false);
        }
        // C/D layout: acc[r] = C[r + 8*half][n0 + (lane&15)]
#pragma unroll
        for (int r = 0; r < 8; ++r)
            Y[(size_t)(m0 + r + half * 8) * FF + n0 + m] = acc[r];
    }
}

// ---------------- out = xw * inv_sqrt^2 (self-loop) + bias ----------------
__global__ void init_out(const float* __restrict__ xw, const float* __restrict__ isq,
                         const float* __restrict__ bias, float* __restrict__ out, long total) {
    long i = (long)blockIdx.x * blockDim.x + threadIdx.x;
    if (i >= total) return;
    int n = (int)(i >> 7), f = (int)(i & 127);
    float s = isq[n];
    out[i] = xw[i] * (s * s) + bias[f];
}

// ---------------- edge scatter: one wave per edge, float4 per lane ----------------
__global__ void edge_scatter(const long long* __restrict__ src, const long long* __restrict__ dst,
                             const float* __restrict__ xw, const float* __restrict__ isq,
                             float* __restrict__ out, int e) {
    int edge = (blockIdx.x * blockDim.x + threadIdx.x) >> 5;
    int lane = threadIdx.x & 31;
    if (edge >= e) return;
    int s = (int)src[edge], d = (int)dst[edge];
    float coef = isq[s] * isq[d];
    float4 v = ((const float4*)(xw + (size_t)s * FF))[lane];
    float* o = out + (size_t)d * FF + lane * 4;
    atomicAdd(o + 0, v.x * coef);
    atomicAdd(o + 1, v.y * coef);
    atomicAdd(o + 2, v.z * coef);
    atomicAdd(o + 3, v.w * coef);
}

__global__ void elu_ip(float* __restrict__ p, long total) {
    long i = (long)blockIdx.x * blockDim.x + threadIdx.x;
    if (i >= total) return;
    float v = p[i];
    p[i] = v > 0.0f ? v : expm1f(v);
}

__global__ void final_copy(const float* __restrict__ h, const float* __restrict__ x,
                           float* __restrict__ out, long total) {
    long i = (long)blockIdx.x * blockDim.x + threadIdx.x;
    if (i >= total) return;
    out[i] = (i < (long)UU * FF) ? h[i] : x[i];
}

extern "C" void kernel_launch(void* const* d_in, const int* in_sizes, int n_in,
                              void* d_out, int out_size, void* d_ws, size_t ws_size,
                              hipStream_t stream) {
    const float*     x   = (const float*)d_in[0];
    const long long* ei  = (const long long*)d_in[1];   // int64 [2, E]
    const float*     W1  = (const float*)d_in[2];
    const float*     b1  = (const float*)d_in[3];
    const float*     W2  = (const float*)d_in[4];
    const float*     b2  = (const float*)d_in[5];
    float*           out = (float*)d_out;

    const long long* srcp = ei;
    const long long* dstp = ei + EE;

    // workspace carve-out (256B aligned): ~102.9 MB
    char* ws = (char*)d_ws;
    size_t off = 0;
    auto carve = [&](size_t bytes) -> void* {
        void* p = ws + off;
        off = (off + bytes + 255) & ~(size_t)255;
        return p;
    };
    float*  isq  = (float*)carve((size_t)NN * 4);
    float*  xw   = (float*)carve((size_t)NN * FF * 4);
    float*  bufA = (float*)carve((size_t)NN * FF * 4);
    __bf16* Wt1  = (__bf16*)carve((size_t)FF * FF * 2);
    __bf16* Wt2  = (__bf16*)carve((size_t)FF * FF * 2);

    const long totalNF = (long)NN * FF;
    const int  blkNF   = (int)((totalNF + 255) / 256);
    const int  blkN    = (NN + 255) / 256;
    const int  blkE    = (EE + 255) / 256;
    const int  blkEw   = (EE * 32 + 255) / 256;                 // one wave per edge
    const int  gemmBlk = (((NN + 15) / 16) + 7) / 8;            // 8 waves per 256-thread block

    // --- normalization ---
    fill_ones<<<blkN, 256, 0, stream>>>(isq, NN);
    deg_count<<<blkE, 256, 0, stream>>>(dstp, isq, EE);
    rsqrt_ip<<<blkN, 256, 0, stream>>>(isq, NN);

    // --- weights to bf16 (transposed) ---
    conv_wt<<<(FF * FF + 255) / 256, 256, 0, stream>>>(W1, Wt1);
    conv_wt<<<(FF * FF + 255) / 256, 256, 0, stream>>>(W2, Wt2);

    // --- layer 1 ---
    gemm_wmma<<<gemmBlk, 256, 0, stream>>>(x, Wt1, xw, NN);
    init_out<<<blkNF, 256, 0, stream>>>(xw, isq, b1, bufA, totalNF);
    edge_scatter<<<blkEw, 256, 0, stream>>>(srcp, dstp, xw, isq, bufA, EE);
    elu_ip<<<blkNF, 256, 0, stream>>>(bufA, totalNF);           // bufA = h1

    // --- layer 2 ---
    gemm_wmma<<<gemmBlk, 256, 0, stream>>>(bufA, Wt2, xw, NN);  // xw = h1 @ W2
    init_out<<<blkNF, 256, 0, stream>>>(xw, isq, b2, bufA, totalNF); // bufA reused (h1 consumed)
    edge_scatter<<<blkEw, 256, 0, stream>>>(srcp, dstp, xw, isq, bufA, EE);
    elu_ip<<<blkNF, 256, 0, stream>>>(bufA, totalNF);           // bufA = h2

    // --- splice: out = [h2[:U], x[U:]] ---
    final_copy<<<blkNF, 256, 0, stream>>>(bufA, x, out, totalNF);
}
